// LSTM_7164005449729
// MI455X (gfx1250) — 8-bit matrix operand order repaired and checked by reading, compiled, not run on hardware
//
#include <hip/hip_runtime.h>
#include <stdint.h>

#define B_    4096
#define T_    1024
#define H_    256
#define C_    10
#define MTILE 16
#define NWAVES 16
#define BLOCK (NWAVES * 32)

typedef __attribute__((ext_vector_type(16))) int   v16i;
typedef __attribute__((ext_vector_type(8)))  float v8f;

#if defined(__has_builtin)
#if __has_builtin(__builtin_amdgcn_cvt_pk_fp8_f32)
#define HAVE_CVT_FP8 1
#endif
#endif

__device__ __forceinline__ unsigned f32_to_fp8_sw(float f) {
  unsigned u = __float_as_uint(f);
  unsigned s = (u >> 31) << 7;
  u &= 0x7fffffffu;
  u += 0x80000u;
  int e = (int)(u >> 23) - 127 + 7;
  unsigned m = (u >> 20) & 7u;
  unsigned r = (e <= 0) ? 0u : (unsigned)((e << 3) | m);
  if (r > 0x7eu) r = 0x7eu;
  return r | s;
}

__device__ __forceinline__ unsigned pack4_fp8(float a, float b, float c, float d) {
#ifdef HAVE_CVT_FP8
  int p = __builtin_amdgcn_cvt_pk_fp8_f32(a, b, 0, false);
  p     = __builtin_amdgcn_cvt_pk_fp8_f32(c, d, p, true);
  return (unsigned)p;
#else
  return f32_to_fp8_sw(a) | (f32_to_fp8_sw(b) << 8)
       | (f32_to_fp8_sw(c) << 16) | (f32_to_fp8_sw(d) << 24);
#endif
}

__device__ __forceinline__ float sig_(float x) {
  float e = __expf(-x);
  return __builtin_amdgcn_rcpf(1.0f + e);
}
__device__ __forceinline__ float tanh_(float x) {
  float e = __expf(-2.0f * x);
  return 1.0f - 2.0f * e * __builtin_amdgcn_rcpf(1.0f + e);
}

__device__ __forceinline__ void async_copy_b32(const void* gsrc, unsigned lds_off) {
  unsigned long long ga = (unsigned long long)(uintptr_t)gsrc;
  asm volatile("global_load_async_to_lds_b32 %0, %1, off"
               :: "v"(lds_off), "v"(ga) : "memory");
}
__device__ __forceinline__ void wait_asynccnt0() {
  asm volatile("s_wait_asynccnt 0" ::: "memory");
}

__global__ __launch_bounds__(BLOCK) void lstm_persistent_fp8(
    const float* __restrict__ x,
    const float* __restrict__ w_gx, const float* __restrict__ w_ix,
    const float* __restrict__ w_fx, const float* __restrict__ w_ox,
    const float* __restrict__ w_gh, const float* __restrict__ w_ih,
    const float* __restrict__ w_fh, const float* __restrict__ w_oh,
    const float* __restrict__ w_ph,
    const float* __restrict__ b_g, const float* __restrict__ b_i,
    const float* __restrict__ b_f, const float* __restrict__ b_o,
    const float* __restrict__ b_p,
    float* __restrict__ out)
{
  extern __shared__ unsigned char smem[];
  float*         xs   = (float*)smem;
  unsigned char* hbuf = smem + 64 * 1024;
  float*         hf   = (float*)smem;

  const int tid  = threadIdx.x;
  const int wave = tid >> 5;
  const int lane = tid & 31;
  const int b0   = blockIdx.x * MTILE;
  const int ncol = wave * 16 + (lane & 15);
  const int mrow = lane & 15;
  const int hk   = (lane < 16) ? 0 : 8;
  const int bk   = (lane < 16) ? 0 : 16;

  {
    const unsigned xs_base = (unsigned)(uintptr_t)xs;
    for (int i = tid; i < MTILE * T_; i += BLOCK) {
      int row = i >> 10, t = i & (T_ - 1);
      async_copy_b32(x + (size_t)(b0 + row) * T_ + t,
                     xs_base + (unsigned)(t * MTILE + row) * 4u);
    }
  }
  for (int i = tid; i < MTILE * H_; i += BLOCK) hbuf[i] = 0;

  const float wxg = w_gx[ncol], wxi = w_ix[ncol], wxf = w_fx[ncol], wxo = w_ox[ncol];
  const float bg  = b_g[ncol],  bi  = b_i[ncol],  bfv = b_f[ncol],  bo  = b_o[ncol];

  v16i bw[8];
  const float* wh[4] = { w_gh, w_ih, w_fh, w_oh };
  #pragma unroll
  for (int g = 0; g < 4; ++g) {
    const float* wp = wh[g] + (size_t)ncol * H_;
    #pragma unroll
    for (int kb = 0; kb < 2; ++kb) {
      union { v16i v; unsigned d[16]; } Bf;
      #pragma unroll
      for (int j = 0; j < 16; ++j) {
        int k = kb * 128 + (j >> 1) * 16 + hk + (j & 1) * 4;
        const float4 w4 = *(const float4*)(wp + k);
        Bf.d[j] = pack4_fp8(w4.x, w4.y, w4.z, w4.w);
      }
      bw[g * 2 + kb] = Bf.v;
    }
  }

  wait_asynccnt0();
  __syncthreads();

  v8f cst = {0.f, 0.f, 0.f, 0.f, 0.f, 0.f, 0.f, 0.f};
  v8f hlast = cst;

  for (int t = 0; t < T_; ++t) {
    const int cur = t & 1, nxt = cur ^ 1;
    const unsigned char* hb = hbuf + cur * (MTILE * H_);

    union { v16i v; unsigned long long q[8]; } A0, A1;
    #pragma unroll
    for (int j = 0; j < 8; ++j) {
      A0.q[j] = *(const unsigned long long*)(hb + mrow * H_ +       j * 16 + hk);
      A1.q[j] = *(const unsigned long long*)(hb + mrow * H_ + 128 + j * 16 + hk);
    }

    const float* xst = xs + t * MTILE + ((lane < 16) ? 0 : 8);
    const float4 xa = *(const float4*)(xst);
    const float4 xb = *(const float4*)(xst + 4);
    float xv[8] = { xa.x, xa.y, xa.z, xa.w, xb.x, xb.y, xb.z, xb.w };

    v8f cg, ci, cf, co;
    #pragma unroll
    for (int j = 0; j < 8; ++j) {
      cg[j] = xv[j] * wxg + bg;  ci[j] = xv[j] * wxi + bi;
      cf[j] = xv[j] * wxf + bfv; co[j] = xv[j] * wxo + bo;
    }

    cg = __builtin_amdgcn_wmma_f32_16x16x128_fp8_fp8(A0.v, bw[0], (short)0, cg, false, false);
    cg = __builtin_amdgcn_wmma_f32_16x16x128_fp8_fp8(A1.v, bw[1], (short)0, cg, false, false);
    ci = __builtin_amdgcn_wmma_f32_16x16x128_fp8_fp8(A0.v, bw[2], (short)0, ci, false, false);
    ci = __builtin_amdgcn_wmma_f32_16x16x128_fp8_fp8(A1.v, bw[3], (short)0, ci, false, false);
    cf = __builtin_amdgcn_wmma_f32_16x16x128_fp8_fp8(A0.v, bw[4], (short)0, cf, false, false);
    cf = __builtin_amdgcn_wmma_f32_16x16x128_fp8_fp8(A1.v, bw[5], (short)0, cf, false, false);
    co = __builtin_amdgcn_wmma_f32_16x16x128_fp8_fp8(A0.v, bw[6], (short)0, co, false, false);
    co = __builtin_amdgcn_wmma_f32_16x16x128_fp8_fp8(A1.v, bw[7], (short)0, co, false, false);

    unsigned char* hn = hbuf + nxt * (MTILE * H_);
    float hv[8];
    #pragma unroll
    for (int j = 0; j < 8; ++j) {
      float gv = tanh_(cg[j]);
      float iv = sig_(ci[j]);
      float fv = sig_(cf[j]);
      float ov = sig_(co[j]);
      float cn = gv * iv + cst[j] * fv;
      cst[j]   = cn;
      hv[j]    = tanh_(cn) * ov;
      hlast[j] = hv[j];
    }
    #pragma unroll
    for (int p = 0; p < 2; ++p) {
      unsigned pk = pack4_fp8(hv[4*p], hv[4*p+1], hv[4*p+2], hv[4*p+3]);
      #pragma unroll
      for (int q = 0; q < 4; ++q) {
        int j = 4 * p + q;
        hn[((lane < 16) ? j : j + 8) * H_ + ncol] = (unsigned char)(pk >> (8 * q));
      }
    }
    __syncthreads();
  }

  #pragma unroll
  for (int j = 0; j < 8; ++j)
    hf[((lane < 16) ? j : j + 8) * H_ + ncol] = hlast[j];
  __syncthreads();
  if (tid < MTILE * C_) {
    int r = tid / C_, cc = tid % C_;
    float acc = b_p[cc];
    const float* wp = w_ph + (size_t)cc * H_;
    for (int k = 0; k < H_; ++k) acc += hf[r * H_ + k] * wp[k];
    out[(size_t)(b0 + r) * C_ + cc] = acc;
  }
}

extern "C" void kernel_launch(void* const* d_in, const int* in_sizes, int n_in,
                              void* d_out, int out_size, void* d_ws, size_t ws_size,
                              hipStream_t stream) {
  const float* x    = (const float*)d_in[0];
  const float* w_gx = (const float*)d_in[1];
  const float* w_ix = (const float*)d_in[2];
  const float* w_fx = (const float*)d_in[3];
  const float* w_ox = (const float*)d_in[4];
  const float* w_gh = (const float*)d_in[5];
  const float* w_ih = (const float*)d_in[6];
  const float* w_fh = (const float*)d_in[7];
  const float* w_oh = (const float*)d_in[8];
  const float* w_ph = (const float*)d_in[9];
  const float* b_g  = (const float*)d_in[10];
  const float* b_i  = (const float*)d_in[11];
  const float* b_f  = (const float*)d_in[12];
  const float* b_o  = (const float*)d_in[13];
  const float* b_p  = (const float*)d_in[14];
  float* out = (float*)d_out;

  const size_t shmem = 64 * 1024 + 8 * 1024;
  dim3 grid(B_ / MTILE), block(BLOCK);
  lstm_persistent_fp8<<<grid, block, shmem, stream>>>(
      x, w_gx, w_ix, w_fx, w_ox, w_gh, w_ih, w_fh, w_oh, w_ph,
      b_g, b_i, b_f, b_o, b_p, out);
}
